// SelfAttention_81406810128473
// MI455X (gfx1250) — compile-verified, matrix operand order checked by reading, not run on hardware
//
#include <hip/hip_runtime.h>
#include <hip/hip_bf16.h>

typedef __attribute__((ext_vector_type(16))) _Float16 v16h;
typedef __attribute__((ext_vector_type(8)))  float    v8f;
typedef __attribute__((ext_vector_type(8)))  _Float16 h8;

#define WMMA_F16(a, b, c) \
  __builtin_amdgcn_wmma_f32_16x16x32_f16(false, (a), false, (b), (short)0, (c), false, false)

// K-index of element e (0..15) of a v16h A/B fragment for half-wave hh (0/1).
__device__ __forceinline__ int kmap(int e, int hh) {
  return ((e >> 3) << 4) + (hh << 3) + (e & 7);
}

// A-fragment: A[m][k] = p[(row0+m)*ld + k0 + kmap(e,hh)], two 16B loads.
__device__ __forceinline__ v16h load_afrag_f16(const _Float16* p, int ld, int row0, int k0) {
  int lane = threadIdx.x & 31;
  int m = lane & 15, hh = lane >> 4;
  const _Float16* r = p + (size_t)(row0 + m) * ld + k0 + (hh << 3);
  h8 lo = *(const h8*)r;
  h8 hi = *(const h8*)(r + 16);
  v16h a;
#pragma unroll
  for (int e = 0; e < 8; ++e) { a[e] = lo[e]; a[e + 8] = hi[e]; }
  return a;
}

// B-fragment from a K-major (row = K) global matrix: strided, used only for
// once-per-block g loads.
__device__ __forceinline__ v16h load_bfrag_f16(const _Float16* p, int ld, int k0, int col0) {
  int lane = threadIdx.x & 31;
  int n = lane & 15, hh = lane >> 4;
  v16h b;
#pragma unroll
  for (int e = 0; e < 16; ++e)
    b[e] = p[(size_t)(k0 + kmap(e, hh)) * ld + col0 + n];
  return b;
}

// B-fragment from an N-major (transposed) LDS pane: row = column n, contiguous
// in K -> two ds_load_b128 per fragment.
__device__ __forceinline__ v16h load_bfrag_lds_t(const _Float16* p, int ldk, int k0) {
  int lane = threadIdx.x & 31;
  int n = lane & 15, hh = lane >> 4;
  const _Float16* r = p + n * ldk + k0 + (hh << 3);
  h8 lo = *(const h8*)r;
  h8 hi = *(const h8*)(r + 16);
  v16h b;
#pragma unroll
  for (int e = 0; e < 8; ++e) { b[e] = lo[e]; b[e + 8] = hi[e]; }
  return b;
}

// ---------------------------------------------------------------------------
// Kernel 0: one-time f32 -> f16 conversion of the stacked weights
// Wc[640][512] = [Wq(64) ; Wk(64) ; Wv(512)].
// ---------------------------------------------------------------------------
__global__ __launch_bounds__(256) void cvtw_kernel(
    const float* __restrict__ Wq, const float* __restrict__ Wk,
    const float* __restrict__ Wv, _Float16* __restrict__ Wc) {
  int fid = (blockIdx.x * 256 + threadIdx.x) * 8;  // over 640*512
  if (fid >= 640 * 512) return;
  int row = fid >> 9, col = fid & 511;
  const float* src = (row < 64)  ? Wq + (size_t)row * 512
                   : (row < 128) ? Wk + (size_t)(row - 64) * 512
                                 : Wv + (size_t)(row - 128) * 512;
  h8 v;
#pragma unroll
  for (int e = 0; e < 8; ++e) v[e] = (_Float16)src[col + e];
  *(h8*)(Wc + fid) = v;
}

// ---------------------------------------------------------------------------
// Kernel 1: fused projections. f = Wq x (stored transposed [N,64] f16),
// g = Wk x ([64,N] f16), h = Wv x ([512,N] f16). One block per (b, 16-n tile);
// x tile staged TRANSPOSED in LDS; per k-step all 5 A-fragments are issued
// before any WMMA consumes them (software pipelined fragment loads).
// ---------------------------------------------------------------------------
__global__ __launch_bounds__(256) void proj_kernel(
    const float* __restrict__ x, const _Float16* __restrict__ Wc,
    _Float16* __restrict__ f_t, _Float16* __restrict__ g_m,
    _Float16* __restrict__ h_m) {
  const int N = 4096, C = 512, XLD = 520;  // pad: bank-stride 4 dwords
  __shared__ _Float16 xl[16 * XLD];        // 16.25 KB

  int b  = blockIdx.y;
  int n0 = blockIdx.x << 4;
  const float* xb = x + (size_t)b * C * N;

  for (int flat = threadIdx.x; flat < C * 16; flat += 256) {
    int c = flat >> 4, col = flat & 15;
    xl[col * XLD + c] = (_Float16)xb[(size_t)c * N + n0 + col];
  }
  __syncthreads();

  int wave = threadIdx.x >> 5, lane = threadIdx.x & 31;
  int m = lane & 15, hh = lane >> 4;

  v8f acc[5] = {v8f{}, v8f{}, v8f{}, v8f{}, v8f{}};
  for (int k0 = 0; k0 < C; k0 += 32) {
    // issue B-frag + all 5 A-frags, then consume: loads retire behind WMMAs
    v16h bx = load_bfrag_lds_t(xl, XLD, k0);
    v16h a[5];
#pragma unroll
    for (int t = 0; t < 5; ++t)
      a[t] = load_afrag_f16(Wc, 512, (wave + (t << 3)) << 4, k0);
#pragma unroll
    for (int t = 0; t < 5; ++t)
      acc[t] = WMMA_F16(a[t], bx, acc[t]);
  }

  _Float16* ftb = f_t + (size_t)b * N * 64;
  _Float16* gmb = g_m + (size_t)b * 64 * N;
  _Float16* hmb = h_m + (size_t)b * C * N;
#pragma unroll
  for (int t = 0; t < 5; ++t) {
    int orow = (wave + (t << 3)) << 4;
    if (orow < 64) {
      // transposed store: f_t[n0+m][orow + 8*hh + r], one b128 per lane
      h8 v;
#pragma unroll
      for (int r = 0; r < 8; ++r) v[r] = (_Float16)acc[t][r];
      *(h8*)(ftb + (size_t)(n0 + m) * 64 + orow + (hh << 3)) = v;
    } else {
      int rel = (orow < 128) ? orow - 64 : orow - 128;
      _Float16* dst = (orow < 128) ? gmb : hmb;
#pragma unroll
      for (int r = 0; r < 8; ++r)
        dst[(size_t)(rel + (hh << 3) + r) * N + n0 + m] = (_Float16)acc[t][r];
    }
  }
}

// ---------------------------------------------------------------------------
// Kernel 2 (pass A): per-column softmax stats over rows i of S = f^T g.
// One wave per 16-j strip; online (max, sum-exp) across 256 i-tiles.
// Next tile's A-fragments are prefetched across the stats VALU work.
// ---------------------------------------------------------------------------
__global__ __launch_bounds__(256) void passA_kernel(
    const _Float16* __restrict__ f_t, const _Float16* __restrict__ g_m,
    float* __restrict__ mbuf, float* __restrict__ lbuf) {
  const int N = 4096;
  int b = blockIdx.y;
  int wave = threadIdx.x >> 5, lane = threadIdx.x & 31;
  int j0 = (blockIdx.x << 7) + (wave << 4);
  const _Float16* fb = f_t + (size_t)b * N * 64;
  const _Float16* gb = g_m + (size_t)b * 64 * N;

  v16h bg0 = load_bfrag_f16(gb, N, 0, j0);   // loaded once, reused 256x
  v16h bg1 = load_bfrag_f16(gb, N, 32, j0);

  v16h a0 = load_afrag_f16(fb, 64, 0, 0);
  v16h a1 = load_afrag_f16(fb, 64, 0, 32);

  float mrun = -3.0e30f, lrun = 0.0f;
  for (int i0 = 0; i0 < N; i0 += 16) {
    v16h c0f = a0, c1f = a1;
    int inx = (i0 + 16 < N) ? i0 + 16 : 0;
    a0 = load_afrag_f16(fb, 64, inx, 0);   // prefetch next tile
    a1 = load_afrag_f16(fb, 64, inx, 32);
    v8f s = {};
    s = WMMA_F16(c0f, bg0, s);
    s = WMMA_F16(c1f, bg1, s);
    float tmax = s[0];
#pragma unroll
    for (int r = 1; r < 8; ++r) tmax = fmaxf(tmax, s[r]);
    tmax = fmaxf(tmax, __shfl_xor(tmax, 16, 32));  // combine row halves
    float tsum = 0.0f;
#pragma unroll
    for (int r = 0; r < 8; ++r) tsum += __expf(s[r] - tmax);
    tsum += __shfl_xor(tsum, 16, 32);
    float mn = fmaxf(mrun, tmax);
    lrun = lrun * __expf(mrun - mn) + tsum * __expf(tmax - mn);
    mrun = mn;
  }
  if (lane < 16) {
    mbuf[(size_t)b * N + j0 + lane] = mrun;
    lbuf[(size_t)b * N + j0 + lane] = lrun;
  }
}

// ---------------------------------------------------------------------------
// Kernel 3 (pass B): o[:,j] = h . softmax-col(S). Block = (b, 16-j strip),
// 8 waves. Each wave computes a distinct 16-i slice of P = exp(S - m_j),
// stored TRANSPOSED in LDS; per K-step all 4 h A-fragments are issued before
// any WMMA consumes them. Epilogue: out = gamma*(acc/l_j) + x.
// ---------------------------------------------------------------------------
__global__ __launch_bounds__(256) void passB_kernel(
    const _Float16* __restrict__ f_t, const _Float16* __restrict__ g_m,
    const _Float16* __restrict__ h_m, const float* __restrict__ mbuf,
    const float* __restrict__ lbuf, const float* __restrict__ x,
    const float* __restrict__ gamma, float* __restrict__ out) {
  const int N = 4096, C = 512, PLD = 136;  // pad: bank-stride 4 dwords
  __shared__ _Float16 Pl[16 * PLD];        // 4.25 KB transposed P panel

  int b = blockIdx.y, j0 = blockIdx.x << 4;
  int wave = threadIdx.x >> 5, lane = threadIdx.x & 31;
  int m = lane & 15, hh = lane >> 4;

  const _Float16* fb = f_t + (size_t)b * N * 64;
  const _Float16* gb = g_m + (size_t)b * 64 * N;
  const _Float16* hb = h_m + (size_t)b * C * N;

  float mj  = mbuf[(size_t)b * N + j0 + m];
  float rin = 1.0f / lbuf[(size_t)b * N + j0 + m];

  v16h bg0 = load_bfrag_f16(gb, N, 0, j0);
  v16h bg1 = load_bfrag_f16(gb, N, 32, j0);

  v8f acc[4] = {v8f{}, v8f{}, v8f{}, v8f{}};

  for (int ib = 0; ib < N; ib += 128) {
    // prefetch next 128-i h panel rows for this wave's c-subtiles
    int ibn = (ib + 128 < N) ? ib + 128 : ib;
#pragma unroll
    for (int t = 0; t < 4; ++t) {
      int c0 = (wave << 6) + (t << 4);
      __builtin_prefetch(hb + (size_t)(c0 + m) * N + ibn, 0, 1);
    }

    // --- this wave's 16-i slice of S -> P (transposed store) ---
    int i0 = ib + (wave << 4);
    v16h a0 = load_afrag_f16(fb, 64, i0, 0);
    v16h a1 = load_afrag_f16(fb, 64, i0, 32);
    v8f s = {};
    s = WMMA_F16(a0, bg0, s);
    s = WMMA_F16(a1, bg1, s);
    h8 pv;
#pragma unroll
    for (int r = 0; r < 8; ++r) pv[r] = (_Float16)__expf(s[r] - mj);
    *(h8*)(Pl + m * PLD + (wave << 4) + (hh << 3)) = pv;
    __syncthreads();

    // --- consume P panel: acc[c-tile] += h_tile x P (K=32 per step) ---
#pragma unroll
    for (int ks = 0; ks < 4; ++ks) {
      v16h bp = load_bfrag_lds_t(Pl, PLD, ks << 5);
      v16h ah[4];
#pragma unroll
      for (int t = 0; t < 4; ++t)
        ah[t] = load_afrag_f16(hb, N, (wave << 6) + (t << 4), ib + (ks << 5));
#pragma unroll
      for (int t = 0; t < 4; ++t)
        acc[t] = WMMA_F16(ah[t], bp, acc[t]);
    }
    __syncthreads();
  }

  float gam = gamma[0];
  const float* xb = x + (size_t)b * C * N;
  float* ob = out + (size_t)b * C * N;
#pragma unroll
  for (int t = 0; t < 4; ++t) {
#pragma unroll
    for (int r = 0; r < 8; ++r) {
      size_t idx = (size_t)((wave << 6) + (t << 4) + (hh << 3) + r) * N + j0 + m;
      ob[idx] = gam * (acc[t][r] * rin) + xb[idx];
    }
  }
}

extern "C" void kernel_launch(void* const* d_in, const int* in_sizes, int n_in,
                              void* d_out, int out_size, void* d_ws, size_t ws_size,
                              hipStream_t stream) {
  const float* x     = (const float*)d_in[0];
  const float* Wq    = (const float*)d_in[1];
  const float* Wk    = (const float*)d_in[2];
  const float* Wv    = (const float*)d_in[3];
  const float* gamma = (const float*)d_in[4];
  float* out = (float*)d_out;

  // Workspace layout (bytes):
  //   [0,   2M)      f_t  : [B][N][64]  f16 (transposed)
  //   [2M,  4M)      g_m  : [B][64][N]  f16
  //   [4M, 20M)      h_m  : [B][512][N] f16
  //   [20M, +64K)    mbuf : [B][N] f32
  //   [.., +64K)     lbuf : [B][N] f32
  //   [.., +640K)    Wc   : [640][512] f16 (stacked converted weights)
  char* ws = (char*)d_ws;
  _Float16* f_t = (_Float16*)(ws);
  _Float16* g_m = (_Float16*)(ws + (2ull << 20));
  _Float16* h_m = (_Float16*)(ws + (4ull << 20));
  float* mbuf   = (float*)(ws + (20ull << 20));
  float* lbuf   = (float*)(ws + (20ull << 20) + (64ull << 10));
  _Float16* Wc  = (_Float16*)(ws + (20ull << 20) + (128ull << 10));

  cvtw_kernel<<<160, 256, 0, stream>>>(Wq, Wk, Wv, Wc);

  dim3 gp(256, 4);  // (n-tile, batch)
  proj_kernel<<<gp, 256, 0, stream>>>(x, Wc, f_t, g_m, h_m);

  dim3 ga(32, 4);   // (128-j group, batch)
  passA_kernel<<<ga, 256, 0, stream>>>(f_t, g_m, mbuf, lbuf);

  dim3 gb(256, 4);  // (16-j strip, batch)
  passB_kernel<<<gb, 256, 0, stream>>>(f_t, g_m, h_m, mbuf, lbuf, x, gamma, out);
}
